// LIIF_64888365908302
// MI455X (gfx1250) — matrix operand order repaired and checked by reading, not compiled, not run on hardware
//
#include <hip/hip_runtime.h>

// ---------------- problem constants ----------------
#define B_    4
#define C_    64
#define H_    128
#define W_    128
#define Q_    65536          // queries per batch image
#define EPS   1e-6f

// ---------------- WMMA fragment types ----------------
typedef _Float16 v16h __attribute__((ext_vector_type(16)));
typedef _Float16 v8h  __attribute__((ext_vector_type(8)));
typedef float    v8f  __attribute__((ext_vector_type(8)));

// One 32x16 (KxN) f16 B-fragment = 512 halves, stored lane-major:
// frag[lane*16 + h]  with  K = (lane>>4)*16 + h,  N = lane&15
#define FRAG_HALVES 512
#define L0_FRAGS (16*3)      // N-tiles x K-chunks (K padded 68->96)
#define LH_FRAGS (16*8)      // 256x256 layers
#define L4_FRAGS (1*8)       // 256x3, N padded to 16
#define L0_OFF 0
#define L1_OFF (L0_OFF + L0_FRAGS*FRAG_HALVES)
#define L2_OFF (L1_OFF + LH_FRAGS*FRAG_HALVES)
#define L3_OFF (L2_OFF + LH_FRAGS*FRAG_HALVES)
#define L4_OFF (L3_OFF + LH_FRAGS*FRAG_HALVES)
#define FRAG_TOTAL (L4_OFF + L4_FRAGS*FRAG_HALVES)   // 225280 halves

#define HALF_FRAG_HALVES (8*8*FRAG_HALVES)           // 8 N-tiles of a hidden layer = 32768 halves

// ---------------- LDS layout (dynamic) ----------------
// [P0: 64KB][P1: 64KB][act: 128 rows x 256 f16 = 64KB][pred 2KB][area 512B]
#define PBUF_BYTES  65536
#define ACT_HALVES  (128*256)
#define PRED_FLOATS (128*4)
#define SMEM_BYTES  (2*PBUF_BYTES + ACT_HALVES*2 + PRED_FLOATS*4 + 128*4)  // 199168 B

// ---------------- async copy: global -> LDS via ASYNCcnt ----------------
// Each loop iteration is one wave-wide async instruction moving 32 lanes x 16B.
__device__ __forceinline__ void async_copy(uint32_t lds_base, const void* src,
                                           int bytes, int tid) {
  int n = bytes >> 4;
  const char* s = (const char*)src;
  for (int i = tid; i < n; i += 256) {
    uint32_t l = lds_base + (uint32_t)i * 16u;
    const char* g = s + (size_t)i * 16u;
    asm volatile("global_load_async_to_lds_b128 %0, %1, off"
                 :: "v"(l), "v"(g) : "memory");
  }
}
#define WAIT_ASYNC_16() asm volatile("s_wait_asynccnt 0x10" ::: "memory")
#define WAIT_ASYNC_2()  asm volatile("s_wait_asynccnt 0x2"  ::: "memory")
#define WAIT_ASYNC_0()  asm volatile("s_wait_asynccnt 0x0"  ::: "memory")

// =====================================================================
// Prep 1: feat [B,C,H,W] -> featT [B,H,W,C]  (makes gathers contiguous)
// =====================================================================
__global__ void liif_feat_transpose(const float* __restrict__ feat,
                                    float* __restrict__ featT) {
  int idx = blockIdx.x * 256 + threadIdx.x;     // over B*C*H*W = 4M
  int x = idx & (W_ - 1);
  int t = idx >> 7;
  int y = t & (H_ - 1);  t >>= 7;
  int c = t & (C_ - 1);  t >>= 6;
  int b = t;
  featT[(size_t)(((b * H_ + y) * W_ + x)) * C_ + c] = feat[idx];
}

// =====================================================================
// Prep 2: f32 weights -> f16 WMMA B-fragments, padded & pre-swizzled
// =====================================================================
__global__ void liif_prep_weights(const float* __restrict__ w0,
                                  const float* __restrict__ w1,
                                  const float* __restrict__ w2,
                                  const float* __restrict__ w3,
                                  const float* __restrict__ w4,
                                  _Float16* __restrict__ frags) {
  int idx = blockIdx.x * 256 + threadIdx.x;
  if (idx >= FRAG_TOTAL) return;
  const float* w; int off, KC, Kreal, Nreal;
  if (idx < L1_OFF)      { w = w0; off = L0_OFF; KC = 3; Kreal = 68;  Nreal = 256; }
  else if (idx < L2_OFF) { w = w1; off = L1_OFF; KC = 8; Kreal = 256; Nreal = 256; }
  else if (idx < L3_OFF) { w = w2; off = L2_OFF; KC = 8; Kreal = 256; Nreal = 256; }
  else if (idx < L4_OFF) { w = w3; off = L3_OFF; KC = 8; Kreal = 256; Nreal = 256; }
  else                   { w = w4; off = L4_OFF; KC = 8; Kreal = 256; Nreal = 3;   }
  int li     = idx - off;
  int f      = li >> 9;          // fragment id = ntile*KC + k
  int within = li & 511;
  int lane   = within >> 4;
  int h      = within & 15;
  int ntile  = f / KC;
  int k      = f - ntile * KC;
  int krow   = k * 32 + (lane >> 4) * 16 + h;
  int ncol   = ntile * 16 + (lane & 15);
  float v = (krow < Kreal && ncol < Nreal) ? w[krow * Nreal + ncol] : 0.0f;
  frags[idx] = (_Float16)v;
}

// =====================================================================
// Wave-level GEMM helpers
// =====================================================================
template <int KC>
__device__ __forceinline__ void load_afrag(const _Float16* act, int wave, int lane,
                                           v16h* aF) {
  const int m  = lane & 15;
  const int hi = lane >> 4;
  const _Float16* arow = act + (size_t)(wave * 16 + m) * 256;
#pragma unroll
  for (int k = 0; k < KC; ++k) {
    v8h lo = *(const v8h*)(arow + k * 32 + hi * 16);       // K = k*32 + {0..7|16..23}
    v8h hh = *(const v8h*)(arow + k * 32 + hi * 16 + 8);   // K = k*32 + {8..15|24..31}
    aF[k] = __builtin_shufflevector(lo, hh, 0,1,2,3,4,5,6,7,8,9,10,11,12,13,14,15);
  }
}

// 8 N-tiles from fragment base `wbase` (fragments ordered [ntile][k]).
template <int KC>
__device__ __forceinline__ void gemm8(const _Float16* wbase,
                                      const float* __restrict__ bias,
                                      int lane, const v16h* aF, v8f* acc, int ntile0) {
  const int m = lane & 15;
#pragma unroll
  for (int n = 0; n < 8; ++n) {
    float bv = bias[(ntile0 + n) * 16 + m];
    v8f c;
#pragma unroll
    for (int i = 0; i < 8; ++i) c[i] = bv;
#pragma unroll
    for (int k = 0; k < KC; ++k) {
      v16h bF = *(const v16h*)(wbase + (size_t)(n * KC + k) * FRAG_HALVES + lane * 16);
      c = __builtin_amdgcn_wmma_f32_16x16x32_f16(false, aF[k], false, bF,
                                                 (short)0, c, false, false);
    }
    acc[n] = c;
  }
}

// ReLU + pack both 8-tile halves back into row-major f16 activations.
__device__ __forceinline__ void store_act16(_Float16* act, int wave, int lane,
                                            const v8f* accLo, const v8f* accHi) {
  const int m  = lane & 15;
  const int hi = lane >> 4;
  _Float16* wact = act + (size_t)(wave * 16) * 256;
#pragma unroll
  for (int n = 0; n < 8; ++n)
#pragma unroll
    for (int r = 0; r < 8; ++r) {
      wact[(hi * 8 + r) * 256 + n * 16 + m]       = (_Float16)fmaxf(accLo[n][r], 0.0f);
      wact[(hi * 8 + r) * 256 + (n + 8) * 16 + m] = (_Float16)fmaxf(accHi[n][r], 0.0f);
    }
}

// Output layer 256 -> 3 (N padded to 16)
__device__ __forceinline__ void mlp_out(_Float16* act, const _Float16* wbuf,
                                        const float* __restrict__ bias,
                                        float* predbuf, int wave, int lane) {
  const int m  = lane & 15;
  const int hi = lane >> 4;
  float bv = (m < 3) ? bias[m] : 0.0f;
  v8f c;
#pragma unroll
  for (int i = 0; i < 8; ++i) c[i] = bv;
  v16h aF[8];
  load_afrag<8>(act, wave, lane, aF);
#pragma unroll
  for (int k = 0; k < 8; ++k) {
    v16h bF = *(const v16h*)(wbuf + (size_t)k * FRAG_HALVES + lane * 16);
    c = __builtin_amdgcn_wmma_f32_16x16x32_f16(false, aF[k], false, bF,
                                               (short)0, c, false, false);
  }
  if (m < 3) {
#pragma unroll
    for (int r = 0; r < 8; ++r)
      predbuf[(wave * 16 + hi * 8 + r) * 4 + m] = c[r];
  }
}

// =====================================================================
// Main kernel: block = 32 queries x 4 ensemble branches = 128 rows,
// 8 waves, each wave owns one 16-row WMMA tile through the whole MLP.
// Weight halves double-buffered in LDS via ASYNCcnt DMA pipeline.
// =====================================================================
__global__ __launch_bounds__(256, 1)
void liif_mlp(const float* __restrict__ featT, const float* __restrict__ coord,
              const float* __restrict__ cell, const _Float16* __restrict__ frags,
              const float* __restrict__ b0, const float* __restrict__ b1,
              const float* __restrict__ b2, const float* __restrict__ b3,
              const float* __restrict__ b4, float* __restrict__ out) {
  extern __shared__ char smem[];
  _Float16* P0  = (_Float16*)smem;
  _Float16* P1  = (_Float16*)(smem + PBUF_BYTES);
  _Float16* act = (_Float16*)(smem + 2 * PBUF_BYTES);
  float* predbuf = (float*)(smem + 2 * PBUF_BYTES + ACT_HALVES * 2);
  float* areabuf = predbuf + PRED_FLOATS;
  const uint32_t ldsP0 = (uint32_t)(uintptr_t)P0;   // LDS aperture: addr[31:0] = LDS byte offset
  const uint32_t ldsP1 = (uint32_t)(uintptr_t)P1;

  const int tid  = threadIdx.x;
  const int lane = tid & 31;
  const int wave = tid >> 5;

  // ---- kick off DMA: L0 frags -> P0 (12/wave), L1 half0 -> P1 (16/wave) ----
  async_copy(ldsP0, frags + L0_OFF, L0_FRAGS * FRAG_HALVES * 2, tid);
  async_copy(ldsP1, frags + L1_OFF, PBUF_BYTES, tid);

  // ---- build 128 input rows (68 real dims, padded to 96) while DMA runs ----
  {
    int r = tid >> 1, part = tid & 1;        // 2 threads per row
    int branch = r >> 5;
    int qg = blockIdx.x * 32 + (r & 31);     // qg == b*Q + q (Q = 65536)
    int b  = qg >> 16;
    float cd0 = coord[(size_t)qg * 2 + 0], cd1 = coord[(size_t)qg * 2 + 1];
    float cl0 = cell[(size_t)qg * 2 + 0],  cl1 = cell[(size_t)qg * 2 + 1];
    float rx = 1.0f / H_, ry = 1.0f / W_;
    float sx = ((branch >> 1) ? rx : -rx) + EPS;
    float sy = ((branch & 1) ? ry : -ry) + EPS;
    float cx = fminf(fmaxf(cd0 + sx, -1.0f + EPS), 1.0f - EPS);
    float cy = fminf(fmaxf(cd1 + sy, -1.0f + EPS), 1.0f - EPS);
    float fiy = fminf(fmaxf(floorf((cx + 1.0f) * (H_ * 0.5f)), 0.0f), (float)(H_ - 1));
    float fix = fminf(fmaxf(floorf((cy + 1.0f) * (W_ * 0.5f)), 0.0f), (float)(W_ - 1));
    int iy = (int)fiy, ix = (int)fix;
    float qc0 = -1.0f + (2.0f * fiy + 1.0f) / H_;
    float qc1 = -1.0f + (2.0f * fix + 1.0f) / W_;
    float rel0 = (cd0 - qc0) * H_, rel1 = (cd1 - qc1) * W_;

    _Float16* row = act + (size_t)r * 256;
    const float4* src =
        (const float4*)(featT + (size_t)((b * H_ + iy) * W_ + ix) * C_ + part * 32);
#pragma unroll
    for (int i = 0; i < 8; ++i) {
      float4 v = src[i];
      row[part * 32 + i * 4 + 0] = (_Float16)v.x;
      row[part * 32 + i * 4 + 1] = (_Float16)v.y;
      row[part * 32 + i * 4 + 2] = (_Float16)v.z;
      row[part * 32 + i * 4 + 3] = (_Float16)v.w;
    }
    if (part) {
      row[64] = (_Float16)rel0;
      row[65] = (_Float16)rel1;
      row[66] = (_Float16)(cl0 * H_);
      row[67] = (_Float16)(cl1 * W_);
#pragma unroll
      for (int i = 68; i < 96; ++i) row[i] = (_Float16)0.0f;
    } else {
      areabuf[r] = fabsf(rel0 * rel1) + 1e-9f;
    }
  }

  // ---- layer 0: all 16 N-tiles resident in P0 (48 KB) ----
  WAIT_ASYNC_16(); __syncthreads();          // L0 batch retired, L1h0 in flight
  {
    v16h aF[3]; v8f accLo[8], accHi[8];
    load_afrag<3>(act, wave, lane, aF);
    gemm8<3>(P0, b0, lane, aF, accLo, 0);
    gemm8<3>(P0 + (size_t)8 * 3 * FRAG_HALVES, b0, lane, aF, accHi, 8);
    store_act16(act, wave, lane, accLo, accHi);
  }
  __syncthreads();                           // P0 free
  async_copy(ldsP0, frags + L1_OFF + HALF_FRAG_HALVES, PBUF_BYTES, tid);   // L1h1
  WAIT_ASYNC_16(); __syncthreads();          // L1h0 ready in P1

  // ---- layer 1 ----
  {
    v16h aF[8]; v8f accLo[8], accHi[8];
    load_afrag<8>(act, wave, lane, aF);
    gemm8<8>(P1, b1, lane, aF, accLo, 0);
    __syncthreads();                         // P1 free
    async_copy(ldsP1, frags + L2_OFF, PBUF_BYTES, tid);                    // L2h0
    WAIT_ASYNC_16(); __syncthreads();        // L1h1 ready in P0
    gemm8<8>(P0, b1, lane, aF, accHi, 8);
    store_act16(act, wave, lane, accLo, accHi);
  }
  __syncthreads();                           // P0 free, act updated
  async_copy(ldsP0, frags + L2_OFF + HALF_FRAG_HALVES, PBUF_BYTES, tid);   // L2h1
  WAIT_ASYNC_16(); __syncthreads();          // L2h0 ready in P1

  // ---- layer 2 ----
  {
    v16h aF[8]; v8f accLo[8], accHi[8];
    load_afrag<8>(act, wave, lane, aF);
    gemm8<8>(P1, b2, lane, aF, accLo, 0);
    __syncthreads();
    async_copy(ldsP1, frags + L3_OFF, PBUF_BYTES, tid);                    // L3h0
    WAIT_ASYNC_16(); __syncthreads();        // L2h1 ready in P0
    gemm8<8>(P0, b2, lane, aF, accHi, 8);
    store_act16(act, wave, lane, accLo, accHi);
  }
  __syncthreads();
  async_copy(ldsP0, frags + L3_OFF + HALF_FRAG_HALVES, PBUF_BYTES, tid);   // L3h1
  WAIT_ASYNC_16(); __syncthreads();          // L3h0 ready in P1

  // ---- layer 3 ----
  {
    v16h aF[8]; v8f accLo[8], accHi[8];
    load_afrag<8>(act, wave, lane, aF);
    gemm8<8>(P1, b3, lane, aF, accLo, 0);
    __syncthreads();
    async_copy(ldsP1, frags + L4_OFF, L4_FRAGS * FRAG_HALVES * 2, tid);    // L4 (2/wave)
    WAIT_ASYNC_2(); __syncthreads();         // L3h1 ready in P0
    gemm8<8>(P0, b3, lane, aF, accHi, 8);
    store_act16(act, wave, lane, accLo, accHi);
  }
  __syncthreads();
  WAIT_ASYNC_0(); __syncthreads();           // L4 frags ready in P1

  // ---- output layer ----
  mlp_out(act, P1, b4, predbuf, wave, lane);
  __syncthreads();

  // ---- local-ensemble combine (area weights diagonally swapped) ----
  if (tid < 96) {
    int i = tid / 3, ch = tid - i * 3;
    int qg = blockIdx.x * 32 + i;
    float a0 = areabuf[i], a1 = areabuf[32 + i], a2 = areabuf[64 + i], a3 = areabuf[96 + i];
    float tot = a0 + a1 + a2 + a3;
    float v = predbuf[(i)      * 4 + ch] * a3
            + predbuf[(32 + i) * 4 + ch] * a2
            + predbuf[(64 + i) * 4 + ch] * a1
            + predbuf[(96 + i) * 4 + ch] * a0;
    out[(size_t)qg * 3 + ch] = v / tot;
  }
}

// =====================================================================
// Host entry
// =====================================================================
extern "C" void kernel_launch(void* const* d_in, const int* in_sizes, int n_in,
                              void* d_out, int out_size, void* d_ws, size_t ws_size,
                              hipStream_t stream) {
  const float* feat  = (const float*)d_in[0];
  const float* coord = (const float*)d_in[1];
  const float* cell  = (const float*)d_in[2];
  const float* w0 = (const float*)d_in[3];  const float* b0 = (const float*)d_in[4];
  const float* w1 = (const float*)d_in[5];  const float* b1 = (const float*)d_in[6];
  const float* w2 = (const float*)d_in[7];  const float* b2 = (const float*)d_in[8];
  const float* w3 = (const float*)d_in[9];  const float* b3 = (const float*)d_in[10];
  const float* w4 = (const float*)d_in[11]; const float* b4 = (const float*)d_in[12];
  float* out = (float*)d_out;

  // workspace: [featT: B*H*W*C f32 = 16 MB][weight fragments: 225280 f16]
  float* featT = (float*)d_ws;
  _Float16* frags = (_Float16*)((char*)d_ws + (size_t)B_ * H_ * W_ * C_ * sizeof(float));

  liif_feat_transpose<<<(B_ * C_ * H_ * W_) / 256, 256, 0, stream>>>(feat, featT);
  liif_prep_weights<<<(FRAG_TOTAL + 255) / 256, 256, 0, stream>>>(w0, w1, w2, w3, w4, frags);

  (void)hipFuncSetAttribute((const void*)liif_mlp,
                            hipFuncAttributeMaxDynamicSharedMemorySize, SMEM_BYTES);
  liif_mlp<<<(B_ * Q_) / 32, 256, SMEM_BYTES, stream>>>(
      featT, coord, cell, frags, b0, b1, b2, b3, b4, out);
}
